// GAT_34531537059968
// MI455X (gfx1250) — matrix operand order repaired and checked by reading, not compiled, not run on hardware
//
#include <hip/hip_runtime.h>
#include <math.h>
#include <cstdint>
#include <cstddef>

// ---------------------------------------------------------------------------
// GAT (3 layers) for MI455X / gfx1250, wave32, WMMA bf16 16x16x32.
// A/B pre-packed into WMMA fragment layout (bf16). B staged in LDS per block;
// each wave computes a 32x64 slab (8 WMMAs per 32-deep k-step).
// ELU fused into the A-pack pass.
// ---------------------------------------------------------------------------

#define N_NODES   100000
#define N_EDGES   1600000
#define IN_DIM    128
#define HIDD      64
#define HEADS     4
#define OUT_DIM   64
#define NEG_SLOPE 0.2f

typedef __attribute__((ext_vector_type(16))) __bf16 v16bf;
typedef __attribute__((ext_vector_type(8)))  float  v8f;

// float -> bf16, round-to-nearest-even, pure integer ops
__device__ __forceinline__ __bf16 f2bf(float f) {
  uint32_t u = __builtin_bit_cast(uint32_t, f);
  uint32_t r = (u + 0x7FFFu + ((u >> 16) & 1u)) >> 16;
  return __builtin_bit_cast(__bf16, (unsigned short)r);
}

#define WMMA_BF16(a, b, c) \
  __builtin_amdgcn_wmma_f32_16x16x32_bf16(false, (a), false, (b), (short)0, (c), false, false)

// ---------------------------------------------------------------------------
// Pack A (M x K fp32, row-major) into WMMA A-fragment layout:
//   pA[(mt*(K/32) + kb)*32 + lane] = v16bf fragment for that lane.
// Lane (hlf=lane>>4, lrow=lane&15): row m = mt*16+lrow,
//   elems 0..7  <-> k = kb*32 + 8*hlf + i
//   elems 8..15 <-> k = kb*32 + 16 + 8*hlf + i
// Optionally applies ELU while converting (post-activation features feed only
// the next GEMM; attention logits use the pre-activation GEMM output).
// ---------------------------------------------------------------------------
__global__ void pack_a_kernel(const float* __restrict__ feat, v16bf* __restrict__ pA,
                              int M, int K, int apply_elu) {
  int idx = blockIdx.x * blockDim.x + threadIdx.x;
  int kt = K >> 5;
  int total = (M >> 4) * kt * 32;
  if (idx >= total) return;
  int lane = idx & 31;
  int t = idx >> 5;
  int mt = t / kt, kb = t - mt * kt;
  int hlf = lane >> 4, lrow = lane & 15;
  const float* row = feat + (size_t)(mt * 16 + lrow) * K + kb * 32 + 8 * hlf;
  v16bf a;
#pragma unroll
  for (int i = 0; i < 8; ++i) {
    float f = row[i];
    if (apply_elu) f = f > 0.f ? f : expm1f(f);
    a[i] = f2bf(f);
  }
#pragma unroll
  for (int i = 0; i < 8; ++i) {
    float f = row[16 + i];
    if (apply_elu) f = f > 0.f ? f : expm1f(f);
    a[8 + i] = f2bf(f);
  }
  pA[idx] = a;
}

// ---------------------------------------------------------------------------
// Pack B (K x N fp32 weights, row-major) into WMMA B-fragment layout:
//   pB[(nt*(K/32) + kb)*32 + lane], nt = 16-column tile index.
// Lane: column n = nt*16 + lrow, elem j<8 <-> k = kb*32 + 8*hlf + j, j>=8 <-> k = kb*32 + 16 + 8*hlf + (j-8).
// ---------------------------------------------------------------------------
__global__ void pack_b_kernel(const float* __restrict__ W, v16bf* __restrict__ pB,
                              int K, int N) {
  int idx = blockIdx.x * blockDim.x + threadIdx.x;
  int kt = K >> 5;
  int total = (N >> 4) * kt * 32;
  if (idx >= total) return;
  int lane = idx & 31;
  int t = idx >> 5;
  int nt = t / kt, kb = t - nt * kt;
  int hlf = lane >> 4, lrow = lane & 15;
  const float* col = W + (size_t)(kb * 32 + 8 * hlf) * N + nt * 16 + lrow;
  v16bf b;
#pragma unroll
  for (int j = 0; j < 16; ++j) b[j] = f2bf(col[(size_t)(j < 8 ? j : j + 8) * N]);
  pB[idx] = b;
}

// ---------------------------------------------------------------------------
// C[M,N] = A*B from packed bf16 fragments.
// Block = 128 threads (4 waves). All waves share one 64-column group whose B
// fragments are staged in LDS once. Each wave computes a 32x64 slab:
// per k-step: 2 A frags (global b128x2) + 4 B frags (ds b128x2) -> 8 WMMAs.
// M%32==0, K%32==0 (kt<=8), N%64==0.
// ---------------------------------------------------------------------------
__global__ void __launch_bounds__(128)
gemm_packed_wmma(const v16bf* __restrict__ pA, const v16bf* __restrict__ pB,
                 float* __restrict__ C, int M, int K, int N) {
  __shared__ v16bf sB[4 * 8 * 32];            // [nt(4)][kb(<=8)][lane(32)], 32 KB
  const int wave = threadIdx.x >> 5;
  const int lane = threadIdx.x & 31;
  const int ng   = N >> 6;                    // 64-column groups
  const int bm   = blockIdx.x / ng;           // 128-row block index
  const int g    = blockIdx.x - bm * ng;
  const int kt   = K >> 5;

  // cooperative stage of this group's B fragments into LDS
  const int tot = 4 * kt * 32;
  for (int j = threadIdx.x; j < tot; j += 128) {
    int nt  = j / (kt * 32);
    int rem = j - nt * (kt * 32);
    sB[nt * (8 * 32) + rem] = pB[(size_t)(g * 4 + nt) * kt * 32 + rem];
  }
  __syncthreads();

  const int mt0 = bm * 8 + wave * 2;          // first of this wave's 2 m-tiles
  if (mt0 * 16 >= M) return;                  // uniform per wave; after barrier

  const v16bf* ap0 = pA + (size_t)mt0 * kt * 32 + lane;
  const v16bf* ap1 = pA + (size_t)(mt0 + 1) * kt * 32 + lane;

  v8f z = {0.f,0.f,0.f,0.f,0.f,0.f,0.f,0.f};
  v8f acc00 = z, acc01 = z, acc02 = z, acc03 = z;
  v8f acc10 = z, acc11 = z, acc12 = z, acc13 = z;

  for (int kb = 0; kb < kt; ++kb) {
    v16bf a0 = ap0[(size_t)kb * 32];
    v16bf a1 = ap1[(size_t)kb * 32];
    v16bf b0 = sB[0 * (8 * 32) + kb * 32 + lane];
    v16bf b1 = sB[1 * (8 * 32) + kb * 32 + lane];
    v16bf b2 = sB[2 * (8 * 32) + kb * 32 + lane];
    v16bf b3 = sB[3 * (8 * 32) + kb * 32 + lane];
    acc00 = WMMA_BF16(a0, b0, acc00);
    acc01 = WMMA_BF16(a0, b1, acc01);
    acc02 = WMMA_BF16(a0, b2, acc02);
    acc03 = WMMA_BF16(a0, b3, acc03);
    acc10 = WMMA_BF16(a1, b0, acc10);
    acc11 = WMMA_BF16(a1, b1, acc11);
    acc12 = WMMA_BF16(a1, b2, acc12);
    acc13 = WMMA_BF16(a1, b3, acc13);
  }

  // C layout: VGPR r <-> row mtile*16 + 8*hlf + r, col = g*64 + nt*16 + lrow
  const int hlf = lane >> 4, lrow = lane & 15;
  float* c0 = C + (size_t)(mt0 * 16 + 8 * hlf) * N + g * 64 + lrow;
  float* c1 = c0 + (size_t)16 * N;
#pragma unroll
  for (int r = 0; r < 8; ++r) {
    float* cr0 = c0 + (size_t)r * N;
    float* cr1 = c1 + (size_t)r * N;
    cr0[0]  = acc00[r];
    cr0[16] = acc01[r];
    cr0[32] = acc02[r];
    cr0[48] = acc03[r];
    cr1[0]  = acc10[r];
    cr1[16] = acc11[r];
    cr1[32] = acc12[r];
    cr1[48] = acc13[r];
  }
}

// ---------------------------------------------------------------------------
// alpha_s[n,h] = sum_d h[n,h,d]*a[h,d];  alpha_d[n,h] = sum_d h[n,h,d]*a[h,D+d]
// ---------------------------------------------------------------------------
__global__ void alpha_kernel(const float* __restrict__ h, const float* __restrict__ a,
                             float* __restrict__ as_, float* __restrict__ ad_,
                             int n, int H, int D) {
  int idx = blockIdx.x * blockDim.x + threadIdx.x;
  if (idx >= n * H) return;
  int node = idx / H, hh = idx - node * H;
  const float* hp  = h + (size_t)node * H * D + (size_t)hh * D;
  const float* asp = a + (size_t)hh * 2 * D;
  const float* adp = asp + D;
  float s = 0.f, d = 0.f;
  for (int i = 0; i < D; ++i) { s = fmaf(hp[i], asp[i], s); d = fmaf(hp[i], adp[i], d); }
  as_[idx] = s;
  ad_[idx] = d;
}

__global__ void fill_kernel(float* __restrict__ p, float v, size_t n) {
  size_t i = (size_t)blockIdx.x * blockDim.x + threadIdx.x;
  if (i < n) p[i] = v;
}

// float atomic max via monotonic int trick (init must be -inf)
__device__ __forceinline__ void atomicMaxFloat(float* addr, float val) {
  int iv = __builtin_bit_cast(int, val);
  if (iv >= 0) atomicMax((int*)addr, iv);
  else         atomicMin((unsigned int*)addr, (unsigned int)iv);
}

__global__ void edge_max_kernel(const int* __restrict__ src, const int* __restrict__ dst,
                                const float* __restrict__ as_, const float* __restrict__ ad_,
                                float* __restrict__ emax, int H) {
  long long idx = (long long)blockIdx.x * blockDim.x + threadIdx.x;
  if (idx >= (long long)N_EDGES * H) return;
  int e  = (int)(idx / H);
  int hh = (int)(idx - (long long)e * H);
  int s = src[e], d = dst[e];
  float v = as_[s * H + hh] + ad_[d * H + hh];
  v = v >= 0.f ? v : NEG_SLOPE * v;
  atomicMaxFloat(&emax[d * H + hh], v);
}

__global__ void fix_max_kernel(float* __restrict__ emax, size_t n) {
  size_t i = (size_t)blockIdx.x * blockDim.x + threadIdx.x;
  if (i >= n) return;
  float v = emax[i];
  if (!isfinite(v)) emax[i] = 0.f;
}

__global__ void edge_expsum_kernel(const int* __restrict__ src, const int* __restrict__ dst,
                                   const float* __restrict__ as_, const float* __restrict__ ad_,
                                   const float* __restrict__ emax,
                                   float* __restrict__ eexp, float* __restrict__ denom, int H) {
  long long idx = (long long)blockIdx.x * blockDim.x + threadIdx.x;
  if (idx >= (long long)N_EDGES * H) return;
  int e  = (int)(idx / H);
  int hh = (int)(idx - (long long)e * H);
  int s = src[e], d = dst[e];
  float v = as_[s * H + hh] + ad_[d * H + hh];
  v = v >= 0.f ? v : NEG_SLOPE * v;
  float ee = expf(v - emax[d * H + hh]);
  eexp[idx] = ee;
  atomicAdd(&denom[d * H + hh], ee);
}

// out[dst] += h[src] * att ; one thread handles 4 consecutive feature dims
__global__ void aggregate_kernel(const int* __restrict__ src, const int* __restrict__ dst,
                                 const float* __restrict__ h,
                                 const float* __restrict__ eexp, const float* __restrict__ denom,
                                 float* __restrict__ out, int H, int D) {
  const int gpe = (H * D) >> 2;     // 4-float groups per edge
  long long idx = (long long)blockIdx.x * blockDim.x + threadIdx.x;
  if (idx >= (long long)N_EDGES * gpe) return;
  int e = (int)(idx / gpe);
  int g = (int)(idx - (long long)e * gpe);
  int hh = (g << 2) / D;
  int s = src[e], d = dst[e];
  float att = eexp[(size_t)e * H + hh] / (denom[d * H + hh] + 1e-16f);
  const float4 hv = *(const float4*)(h + (size_t)s * H * D + (g << 2));
  float* op = out + (size_t)d * H * D + (g << 2);
  atomicAdd(&op[0], hv.x * att);
  atomicAdd(&op[1], hv.y * att);
  atomicAdd(&op[2], hv.z * att);
  atomicAdd(&op[3], hv.w * att);
}

// ---------------------------------------------------------------------------
// Host-side layer driver (all launches on `stream`; graph-capture safe)
// ---------------------------------------------------------------------------
static inline int cdiv(long long a, int b) { return (int)((a + b - 1) / b); }

static void run_gat_layer(const float* feat_in, int K, bool elu_in,
                          const float* W, const float* a, int H, int D,
                          const int* src, const int* dst,
                          v16bf* pA, v16bf* pB,
                          float* hlin, float* out,
                          float* as_, float* ad_, float* emax, float* denom, float* eexp,
                          hipStream_t stream) {
  const int HD = H * D;
  // 1) pack A (with fused ELU for layers 1/2) and B into WMMA fragments
  const int na = (N_NODES / 16) * (K / 32) * 32;
  pack_a_kernel<<<cdiv(na, 256), 256, 0, stream>>>(feat_in, pA, N_NODES, K, elu_in ? 1 : 0);
  const int nb = (HD / 16) * (K / 32) * 32;
  pack_b_kernel<<<cdiv(nb, 256), 256, 0, stream>>>(W, pB, K, HD);
  // 2) h = feat @ W  (WMMA; 32x64 slab per wave, B staged in LDS)
  const int mblocks = cdiv(N_NODES, 128);
  gemm_packed_wmma<<<mblocks * (HD / 64), 128, 0, stream>>>(pA, pB, hlin, N_NODES, K, HD);
  // 3) attention logits per node/head
  const int nh = N_NODES * H;
  alpha_kernel<<<cdiv(nh, 256), 256, 0, stream>>>(hlin, a, as_, ad_, N_NODES, H, D);
  // 4) init reductions + output
  fill_kernel<<<cdiv(nh, 256), 256, 0, stream>>>(emax, -INFINITY, (size_t)nh);
  fill_kernel<<<cdiv(nh, 256), 256, 0, stream>>>(denom, 0.f, (size_t)nh);
  const size_t nout = (size_t)N_NODES * HD;
  fill_kernel<<<cdiv((long long)nout, 256), 256, 0, stream>>>(out, 0.f, nout);
  // 5) segment softmax over dst
  const long long eh = (long long)N_EDGES * H;
  edge_max_kernel<<<cdiv(eh, 256), 256, 0, stream>>>(src, dst, as_, ad_, emax, H);
  fix_max_kernel<<<cdiv(nh, 256), 256, 0, stream>>>(emax, (size_t)nh);
  edge_expsum_kernel<<<cdiv(eh, 256), 256, 0, stream>>>(src, dst, as_, ad_, emax, eexp, denom, H);
  // 6) weighted scatter-add aggregation
  const long long tot = (long long)N_EDGES * (HD >> 2);
  aggregate_kernel<<<cdiv(tot, 256), 256, 0, stream>>>(src, dst, hlin, eexp, denom, out, H, D);
}

extern "C" void kernel_launch(void* const* d_in, const int* in_sizes, int n_in,
                              void* d_out, int out_size, void* d_ws, size_t ws_size,
                              hipStream_t stream) {
  (void)in_sizes; (void)n_in; (void)out_size; (void)ws_size;
  const float* x  = (const float*)d_in[0];
  const int*   ei = (const int*)d_in[1];
  const int*   src = ei;
  const int*   dst = ei + N_EDGES;
  const float* W0 = (const float*)d_in[2];
  const float* a0 = (const float*)d_in[3];
  const float* W1 = (const float*)d_in[4];
  const float* a1 = (const float*)d_in[5];
  const float* W2 = (const float*)d_in[6];
  const float* a2 = (const float*)d_in[7];
  float* outp = (float*)d_out;

  // workspace layout (float units; every block is a multiple of 32 bytes)
  float* ws    = (float*)d_ws;
  float* bufA  = ws;                                   // N*256  (GEMM out / h)
  float* bufB  = bufA + (size_t)N_NODES * 256;         // N*256  (agg target / next feat)
  float* as_   = bufB + (size_t)N_NODES * 256;         // N*HEADS
  float* ad_   = as_  + (size_t)N_NODES * HEADS;       // N*HEADS
  float* emax  = ad_  + (size_t)N_NODES * HEADS;       // N*HEADS
  float* denom = emax + (size_t)N_NODES * HEADS;       // N*HEADS
  float* eexp  = denom + (size_t)N_NODES * HEADS;      // E*HEADS
  float* pAf   = eexp + (size_t)N_EDGES * HEADS;       // packed A: (N/16)*(256/32)*32 v16bf
  v16bf* pA    = (v16bf*)pAf;
  v16bf* pB    = pA + (size_t)(N_NODES / 16) * (256 / 32) * 32;  // packed B: <=4096 v16bf

  // Layer 0: x(N,128) -> bufB(N,256)
  run_gat_layer(x, IN_DIM, false, W0, a0, HEADS, HIDD, src, dst,
                pA, pB, bufA, bufB, as_, ad_, emax, denom, eexp, stream);
  // Layer 1: elu(bufB)(N,256) -> bufB(N,256) via bufA
  run_gat_layer(bufB, HEADS * HIDD, true, W1, a1, HEADS, HIDD, src, dst,
                pA, pB, bufA, bufB, as_, ad_, emax, denom, eexp, stream);
  // Layer 2: elu(bufB)(N,256) -> d_out(N,64); 1 head, mean over 1 head == identity
  run_gat_layer(bufB, HEADS * HIDD, true, W2, a2, 1, OUT_DIM, src, dst,
                pA, pB, bufA, outp, as_, ad_, emax, denom, eexp, stream);
}
